// GWNNLayer_13383118094872
// MI455X (gfx1250) — matrix operand order repaired and checked by reading, not compiled, not run on hardware
//
#include <hip/hip_runtime.h>

// CDNA5 / gfx1250: wave32, WMMA 16x16x32 bf16 with f32 accumulate.
typedef __attribute__((ext_vector_type(16))) __bf16 v16bf;
typedef __attribute__((ext_vector_type(8)))  __bf16 v8bf;
typedef __attribute__((ext_vector_type(8)))  float  v8f;

#define M_TOTAL 8192
#define N_TOTAL 256

// ---------------------------------------------------------------------------
// Tiny prep kernel: Wt[n][k] = bf16(W[k][n])   (256x256, negligible cost)
// ---------------------------------------------------------------------------
__global__ void k_transpose_w(const float* __restrict__ W, __bf16* __restrict__ Wt) {
    int k = blockIdx.x;      // 0..255
    int n = threadIdx.x;     // 0..255
    Wt[(size_t)n * 256 + k] = (__bf16)W[(size_t)k * 256 + n];
}

// ---------------------------------------------------------------------------
// GEMM: C[M=8192, N=256] = A[f32, MxK row-major] @ B (given as Bt[n][k] bf16)
// Wave tile 64x64 (4x4 WMMA accumulators), block = 8 waves -> 128x256 tile.
// MODE 0: store C transposed as bf16  -> Ct[n][m]            (feeds next GEMM)
// MODE 1: store filt[m]*C transposed  -> Ct[n][m] bf16       (fused diag scale)
// MODE 2: store C row-major f32       -> out[m*256+n]        (final output)
// ---------------------------------------------------------------------------
template <int MODE>
__global__ __launch_bounds__(256)
void k_gemm_bf16(const float* __restrict__ A,
                 const __bf16* __restrict__ Bt,
                 int K,
                 void* __restrict__ Cout,
                 const float* __restrict__ filt) {
    const int lane = threadIdx.x & 31;
    const int wave = threadIdx.x >> 5;     // 0..7
    const int wm   = wave & 1;             // M-wave 0..1
    const int wn   = wave >> 1;            // N-wave 0..3
    const int r    = lane & 15;            // row (A) / col (B,C) within 16-tile
    const int h    = lane >> 4;            // lane half

    const int Mb = blockIdx.x * 128 + wm * 64;
    const int Nb = wn * 64;

    v8f acc[4][4];
#pragma unroll
    for (int i = 0; i < 4; ++i)
#pragma unroll
        for (int j = 0; j < 4; ++j)
            acc[i][j] = (v8f)0.0f;

    for (int kb = 0; kb < K; kb += 32) {
        // B fragments: 32x16 bf16, lane layout K = kb + 16*h + e  (contiguous 32B)
        v16bf bfrag[4];
#pragma unroll
        for (int j = 0; j < 4; ++j) {
            const __bf16* bp = Bt + (size_t)(Nb + 16 * j + r) * K + (kb + 16 * h);
            bfrag[j] = *(const v16bf*)bp;
        }
        // A fragments: 16x32 bf16, lane layout:
        //   K = kb + 16h + e
#pragma unroll
        for (int i = 0; i < 4; ++i) {
            const float* ap = A + (size_t)(Mb + 16 * i + r) * K + (kb + 16 * h);
            v8f c1 = *(const v8f*)ap;          // K = kb+16h   .. +8
            v8f c2 = *(const v8f*)(ap + 8);    // K = kb+16h+8 .. +8
            v16bf a;
#pragma unroll
            for (int e = 0; e < 8; ++e) {
                a[e]     = (__bf16)c1[e];
                a[e + 8] = (__bf16)c2[e];
            }
#pragma unroll
            for (int j = 0; j < 4; ++j) {
                acc[i][j] = __builtin_amdgcn_wmma_f32_16x16x32_bf16(
                    false, a, false, bfrag[j], (short)0, acc[i][j], false, false);
            }
        }
    }

    // C/D fragment layout: lane holds column n = r; VGPR g holds row m = g + 8h.
    if (MODE == 2) {
        float* out = (float*)Cout;
#pragma unroll
        for (int i = 0; i < 4; ++i) {
            const int m0 = Mb + 16 * i + 8 * h;
#pragma unroll
            for (int j = 0; j < 4; ++j) {
                const int n = Nb + 16 * j + r;
#pragma unroll
                for (int g = 0; g < 8; ++g)
                    out[(size_t)(m0 + g) * N_TOTAL + n] = acc[i][j][g];
            }
        }
    } else {
        __bf16* out = (__bf16*)Cout;   // Ct[n][m], ld = M_TOTAL
        float fs[4][8];
#pragma unroll
        for (int i = 0; i < 4; ++i) {
            const int m0 = Mb + 16 * i + 8 * h;
#pragma unroll
            for (int g = 0; g < 8; ++g)
                fs[i][g] = (MODE == 1) ? filt[m0 + g] : 1.0f;
        }
#pragma unroll
        for (int i = 0; i < 4; ++i) {
            const int m0 = Mb + 16 * i + 8 * h;   // multiple of 8 -> 16B aligned
#pragma unroll
            for (int j = 0; j < 4; ++j) {
                const int n = Nb + 16 * j + r;
                v8bf pk;
#pragma unroll
                for (int g = 0; g < 8; ++g)
                    pk[g] = (__bf16)(acc[i][j][g] * fs[i][g]);
                *(v8bf*)(out + (size_t)n * M_TOTAL + m0) = pk;   // one b128 store
            }
        }
    }
}

// ---------------------------------------------------------------------------
// Host launcher
// d_in: 0=features[8192,256] 1=wavelets[8192,8192] 2=wavelets_inv[8192,8192]
//       3=weight_matrix[256,256] 4=filt[8192]      (all f32)
// d_out: f32 [8192,256]
// Workspace: Tt (4MB bf16) | Ft (4MB bf16) | Wt (128KB bf16)
// ---------------------------------------------------------------------------
extern "C" void kernel_launch(void* const* d_in, const int* in_sizes, int n_in,
                              void* d_out, int out_size, void* d_ws, size_t ws_size,
                              hipStream_t stream) {
    const float* features     = (const float*)d_in[0];
    const float* wavelets     = (const float*)d_in[1];
    const float* wavelets_inv = (const float*)d_in[2];
    const float* W            = (const float*)d_in[3];
    const float* filt         = (const float*)d_in[4];
    float*       out          = (float*)d_out;

    char* ws = (char*)d_ws;
    const size_t bigMat = (size_t)N_TOTAL * M_TOTAL * sizeof(__bf16);   // 4 MB
    __bf16* Tt = (__bf16*)(ws);                     // T^T  [256][8192]
    __bf16* Ft = (__bf16*)(ws + bigMat);            // F^T  [256][8192]
    __bf16* Wt = (__bf16*)(ws + 2 * bigMat);        // W^T  [256][256]

    // 1) W^T in bf16
    k_transpose_w<<<256, 256, 0, stream>>>(W, Wt);

    // 2) T = features @ W            -> Tt (bf16, transposed)
    k_gemm_bf16<0><<<M_TOTAL / 128, 256, 0, stream>>>(features, Wt, 256, (void*)Tt, nullptr);

    // 3) F = diag(filt) @ (wavelets_inv @ T)   -> Ft (bf16, transposed, scale fused)
    k_gemm_bf16<1><<<M_TOTAL / 128, 256, 0, stream>>>(wavelets_inv, Tt, M_TOTAL, (void*)Ft, filt);

    // 4) out = wavelets @ F          -> f32 row-major
    k_gemm_bf16<2><<<M_TOTAL / 128, 256, 0, stream>>>(wavelets, Ft, M_TOTAL, (void*)out, nullptr);
}
